// VarianceAdaptor_4621384810520
// MI455X (gfx1250) — compile-verified, matrix operand order checked by reading, not run on hardware
//
#include <hip/hip_runtime.h>

// ---------------------------------------------------------------------------
// VarianceAdaptor for MI455X (gfx1250, wave32).
// Conv1d layers = im2col GEMMs on v_wmma_f32_16x16x32_bf16 (f32 accumulate),
// fused with bias+ReLU+LayerNorm per 32-row tile in LDS. bf16 activations are
// staged into LDS with the Tensor Data Mover (tensor_load_to_lds + TENSORcnt)
// when available; f32 activations are converted in-flight.
// ---------------------------------------------------------------------------

typedef __attribute__((ext_vector_type(8)))  __bf16 bf16x8;
typedef __attribute__((ext_vector_type(16))) __bf16 bf16x16;
typedef __attribute__((ext_vector_type(8)))  float  f32x8;
typedef __attribute__((ext_vector_type(4)))  unsigned int u32x4v;
typedef __attribute__((ext_vector_type(4)))  int i32x4v;
typedef __attribute__((ext_vector_type(8)))  int i32x8v;

#define B_SZ   32
#define T_SZ   256
#define M_SZ   2048
#define C_SZ   256
#define F_SZ   256
#define KKTOT  768          // 3 * 256 im2col K
#define NBQ    255          // boundaries per quantizer
#define TROWS  32           // output rows per block
#define HROWS  (TROWS + 2)  // staged rows incl. conv halo

#if defined(__has_builtin)
#if __has_builtin(__builtin_amdgcn_tensor_load_to_lds) && __has_builtin(__builtin_amdgcn_s_wait_tensorcnt)
#define HAVE_TDM 1
#endif
#endif
#ifndef HAVE_TDM
#define HAVE_TDM 0
#endif

__device__ __forceinline__ unsigned short f2bf(float f) {
    unsigned int u = __float_as_uint(f);
    u += 0x7FFFu + ((u >> 16) & 1u);          // round-to-nearest-even
    return (unsigned short)(u >> 16);
}
__device__ __forceinline__ float bf2f(unsigned short h) {
    return __uint_as_float(((unsigned int)h) << 16);
}

#if HAVE_TDM
// 2-D TDM load: nrows x 256 bf16 tile, row stride 256 elements, -> LDS.
__device__ __forceinline__ void tdm_load_rows_bf16(unsigned lds_off_bytes,
                                                   const unsigned short* gsrc,
                                                   int nrows) {
    const unsigned long long ga = (unsigned long long)gsrc;
    u32x4v g0;
    g0[0] = 1u;                                           // count=1, user D#
    g0[1] = lds_off_bytes;                                // lds_addr
    g0[2] = (unsigned)(ga & 0xFFFFFFFFu);                 // global_addr lo
    g0[3] = (unsigned)((ga >> 32) & 0x1FFFFFFu) | (2u << 30);  // addr hi | type=2
    i32x8v g1;
    g1[0] = (int)(1u << 16);                              // data_size = 2 bytes
    g1[1] = (int)(((unsigned)C_SZ & 0xFFFFu) << 16);      // tensor_dim0[15:0]
    g1[2] = (int)(((unsigned)nrows & 0xFFFFu) << 16);     // tensor_dim1[15:0]
    g1[3] = (int)(((unsigned)C_SZ & 0xFFFFu) << 16);      // tile_dim0 = 256
    g1[4] = (int)((unsigned)nrows & 0xFFFFu);             // tile_dim1 = nrows
    g1[5] = C_SZ;                                         // tensor_dim0_stride lo
    g1[6] = 0;
    g1[7] = 0;
    const i32x4v z4 = {0, 0, 0, 0};
#if __clang_major__ >= 23
    const i32x8v z8 = {0, 0, 0, 0, 0, 0, 0, 0};
    __builtin_amdgcn_tensor_load_to_lds(g0, g1, z4, z4, z8, 0);
#else
    __builtin_amdgcn_tensor_load_to_lds(g0, g1, z4, z4, 0);
#endif
}
#endif

// ---------------------------------------------------------------------------
// Repack conv weights [F, C, K] f32 -> bf16 [F][k*C + c] (im2col column order)
// ---------------------------------------------------------------------------
__global__ __launch_bounds__(256)
void repack_w_kernel(const float* __restrict__ w, unsigned short* __restrict__ wp) {
    const int f = blockIdx.x;
    const int c = threadIdx.x;
#pragma unroll
    for (int k = 0; k < 3; ++k)
        wp[(long long)f * KKTOT + k * C_SZ + c] = f2bf(w[(long long)f * KKTOT + c * 3 + k]);
}

// ---------------------------------------------------------------------------
// Per-batch duration cumsum + mel_len
// ---------------------------------------------------------------------------
__global__ void cumsum_kernel(const int* __restrict__ dur, int* __restrict__ cs,
                              float* __restrict__ mel_len_out,
                              const int* __restrict__ max_len_ptr) {
    const int b = threadIdx.x;
    if (b >= B_SZ) return;
    int s = 0;
    for (int t = 0; t < T_SZ; ++t) { s += dur[b * T_SZ + t]; cs[b * T_SZ + t] = s; }
    const int ml = max_len_ptr[0];
    mel_len_out[b] = (float)(s < ml ? s : ml);
}

// ---------------------------------------------------------------------------
// Length regulation: out[b,m,:] = valid ? x[b, searchsorted(cs, m, right), :] : 0
// ---------------------------------------------------------------------------
__global__ __launch_bounds__(256)
void length_regulate_kernel(const float* __restrict__ x, const int* __restrict__ cs,
                            float* __restrict__ out) {
    __shared__ int sidx[16];
    const int tilesPerB = M_SZ / 16;
    const int b  = blockIdx.x / tilesPerB;
    const int m0 = (blockIdx.x % tilesPerB) * 16;
    if (threadIdx.x < 16) {
        const int t = m0 + threadIdx.x;
        const int* c = &cs[b * T_SZ];
        const int total = c[T_SZ - 1];
        int lo = 0, hi = T_SZ;
        while (lo < hi) { int mid = (lo + hi) >> 1; if (c[mid] <= t) lo = mid + 1; else hi = mid; }
        if (lo > T_SZ - 1) lo = T_SZ - 1;
        sidx[threadIdx.x] = (t < total) ? lo : -1;
    }
    __syncthreads();
    const int r  = threadIdx.x >> 4;
    const int c0 = (threadIdx.x & 15) * 16;
    const int idx = sidx[r];
    float4* op = reinterpret_cast<float4*>(&out[(((long long)b * M_SZ) + m0 + r) * C_SZ + c0]);
    if (idx < 0) {
        float4 z = make_float4(0.f, 0.f, 0.f, 0.f);
#pragma unroll
        for (int i = 0; i < 4; ++i) op[i] = z;
    } else {
        const float4* ip = reinterpret_cast<const float4*>(&x[(((long long)b * T_SZ) + idx) * C_SZ + c0]);
#pragma unroll
        for (int i = 0; i < 4; ++i) op[i] = ip[i];
    }
}

// ---------------------------------------------------------------------------
// Fused Conv1d(K=3,'same') + bias + ReLU + LayerNorm.
// Block = 32 rows x 256 cols. 8 waves; each wave owns a 2x2 grid of 16x16 f32
// accumulators (rows {0,16}+.., cols wave*32 + {0,16}); K loop = 24 chunks of
// 32 -> 96 v_wmma per wave, with A and B fragments each reused twice.
// ---------------------------------------------------------------------------
__global__ __launch_bounds__(256)
void conv_relu_ln_kernel(const float* __restrict__ xin_f32,
                         const unsigned short* __restrict__ xin_bf16,
                         const unsigned short* __restrict__ wpack,   // [256][768] bf16
                         const float* __restrict__ bias,
                         const float* __restrict__ gamma,
                         const float* __restrict__ beta,
                         unsigned short* __restrict__ out,           // [rows][256] bf16
                         int L) {
    __shared__ unsigned short sx[HROWS * C_SZ];  // rows t0-1 .. t0+32 (halo)
    __shared__ float ty[TROWS * F_SZ];           // post-ReLU f32 tile
    __shared__ float red_s[TROWS * 8];
    __shared__ float red_q[TROWS * 8];
    __shared__ float s_mean[TROWS], s_rstd[TROWS];

    const int tid  = threadIdx.x;
    const int lane = tid & 31;
    const int wave = tid >> 5;

    const int tilesPerSeq = L / TROWS;
    const int b  = blockIdx.x / tilesPerSeq;
    const int t0 = (blockIdx.x % tilesPerSeq) * TROWS;
    const long long rowBase = (long long)b * L + t0;

    // ---- Stage activations (with conv halo) as bf16 into LDS ----
    if (xin_f32) {
        for (int i = tid; i < HROWS * C_SZ; i += 256) {
            const int r = i >> 8;
            const int c = i & 255;
            const int t = t0 + r - 1;
            unsigned short v = 0;
            if (t >= 0 && t < L) v = f2bf(xin_f32[((long long)b * L + t) * C_SZ + c]);
            sx[i] = v;
        }
    } else {
#if HAVE_TDM
        // Zero halo rows that fall outside the sequence, then one TDM 2-D tile
        // load for the in-range window; wave 0 issues and waits on TENSORcnt.
        if (t0 == 0)
            for (int i = tid; i < C_SZ; i += 256) sx[i] = 0;
        if (t0 + TROWS == L)
            for (int i = tid; i < C_SZ; i += 256) sx[(HROWS - 1) * C_SZ + i] = 0;
        const int tstart  = (t0 == 0) ? 0 : (t0 - 1);
        const int tend    = (t0 + TROWS + 1 < L) ? (t0 + TROWS + 1) : L;
        const int nrows   = tend - tstart;
        const int ldsRow0 = tstart - (t0 - 1);       // 0 or 1
        if (tid == 0) {
            const unsigned lds_off =
                (unsigned)(unsigned long long)(&sx[0]) + (unsigned)(ldsRow0 * C_SZ * 2);
            tdm_load_rows_bf16(lds_off,
                               &xin_bf16[((long long)b * L + tstart) * C_SZ], nrows);
            __builtin_amdgcn_s_wait_tensorcnt(0);
        }
#else
        for (int i = tid; i < HROWS * C_SZ; i += 256) {
            const int r = i >> 8;
            const int c = i & 255;
            const int t = t0 + r - 1;
            unsigned short v = 0;
            if (t >= 0 && t < L) v = xin_bf16[((long long)b * L + t) * C_SZ + c];
            sx[i] = v;
        }
#endif
    }
    __syncthreads();

    // ---- WMMA GEMM ----
    const int n0    = wave * 32;
    const int kbase = (lane < 16) ? 0 : 8;      // ISA 16-bit A/B lane layout
    const int mrow  = lane & 15;
    f32x8 acc00 = {0.f, 0.f, 0.f, 0.f, 0.f, 0.f, 0.f, 0.f};
    f32x8 acc01 = acc00, acc10 = acc00, acc11 = acc00;

    for (int j = 0; j < 24; ++j) {
        const int k  = j >> 3;                  // conv tap 0..2
        const int c0 = (j & 7) * 32;            // channel offset
        union { bf16x16 v; bf16x8 h[2]; } a0, a1, b0, b1;
        const unsigned short* ap0 = &sx[(mrow + k) * C_SZ + c0 + kbase];
        const unsigned short* ap1 = &sx[(mrow + 16 + k) * C_SZ + c0 + kbase];
        a0.h[0] = *reinterpret_cast<const bf16x8*>(ap0);
        a0.h[1] = *reinterpret_cast<const bf16x8*>(ap0 + 16);
        a1.h[0] = *reinterpret_cast<const bf16x8*>(ap1);
        a1.h[1] = *reinterpret_cast<const bf16x8*>(ap1 + 16);
        const unsigned short* bp0 = &wpack[(long long)(n0 + mrow) * KKTOT + j * 32 + kbase];
        const unsigned short* bp1 = &wpack[(long long)(n0 + 16 + mrow) * KKTOT + j * 32 + kbase];
        b0.h[0] = *reinterpret_cast<const bf16x8*>(bp0);
        b0.h[1] = *reinterpret_cast<const bf16x8*>(bp0 + 16);
        b1.h[0] = *reinterpret_cast<const bf16x8*>(bp1);
        b1.h[1] = *reinterpret_cast<const bf16x8*>(bp1 + 16);
        if (j + 1 < 24) {                       // warm next weight chunk
            __builtin_prefetch(bp0 + 32, 0, 1);
            __builtin_prefetch(bp1 + 32, 0, 1);
        }
        acc00 = __builtin_amdgcn_wmma_f32_16x16x32_bf16(false, a0.v, false, b0.v,
                                                        (short)0, acc00, false, false);
        acc01 = __builtin_amdgcn_wmma_f32_16x16x32_bf16(false, a0.v, false, b1.v,
                                                        (short)0, acc01, false, false);
        acc10 = __builtin_amdgcn_wmma_f32_16x16x32_bf16(false, a1.v, false, b0.v,
                                                        (short)0, acc10, false, false);
        acc11 = __builtin_amdgcn_wmma_f32_16x16x32_bf16(false, a1.v, false, b1.v,
                                                        (short)0, acc11, false, false);
    }

    // bias + ReLU -> LDS f32 tile (C/D layout: lane%16 = col, lane/16 = M half)
    {
        const int rb   = (lane >> 4) * 8;
        const int col0 = n0 + (lane & 15);
        const int col1 = col0 + 16;
        const float bv0 = bias[col0], bv1 = bias[col1];
#pragma unroll
        for (int i = 0; i < 8; ++i) {
            ty[(rb + i) * F_SZ + col0]        = fmaxf(acc00[i] + bv0, 0.f);
            ty[(rb + i) * F_SZ + col1]        = fmaxf(acc01[i] + bv1, 0.f);
            ty[(rb + 16 + i) * F_SZ + col0]   = fmaxf(acc10[i] + bv0, 0.f);
            ty[(rb + 16 + i) * F_SZ + col1]   = fmaxf(acc11[i] + bv1, 0.f);
        }
    }
    __syncthreads();

    // LayerNorm over F=256 per row: two-stage reduction (32 rows x 8 segments).
    {
        const int r = tid >> 3, seg = tid & 7;
        float s = 0.f, q = 0.f;
#pragma unroll
        for (int jj = 0; jj < 32; ++jj) {
            const float v = ty[r * F_SZ + seg * 32 + jj];
            s += v; q += v * v;
        }
        red_s[r * 8 + seg] = s;
        red_q[r * 8 + seg] = q;
    }
    __syncthreads();
    if (tid < TROWS) {
        float s = 0.f, q = 0.f;
#pragma unroll
        for (int jj = 0; jj < 8; ++jj) { s += red_s[tid * 8 + jj]; q += red_q[tid * 8 + jj]; }
        const float mean = s * (1.f / 256.f);
        const float var  = q * (1.f / 256.f) - mean * mean;
        s_mean[tid] = mean;
        s_rstd[tid] = rsqrtf(var + 1e-5f);
    }
    __syncthreads();
    {
        const int r = tid >> 3, seg = tid & 7;
        const float mean = s_mean[r], rstd = s_rstd[r];
        const long long obase = (rowBase + r) * F_SZ;
#pragma unroll
        for (int jj = 0; jj < 32; ++jj) {
            const int c = seg * 32 + jj;
            const float v = (ty[r * F_SZ + c] - mean) * rstd * gamma[c] + beta[c];
            out[obase + c] = f2bf(v);
        }
    }
}

// ---------------------------------------------------------------------------
// Linear(F->1) + ReLU: one wave per row, coalesced 16B loads, shuffle reduce.
// ---------------------------------------------------------------------------
__global__ __launch_bounds__(256)
void linear_relu_kernel(const unsigned short* __restrict__ h,
                        const float* __restrict__ wl, const float* __restrict__ bl,
                        float* __restrict__ out, int rows) {
    __shared__ float wls[F_SZ];
    wls[threadIdx.x] = wl[threadIdx.x];
    __syncthreads();
    const int lane = threadIdx.x & 31;
    const int wave = threadIdx.x >> 5;
    const int row  = blockIdx.x * 8 + wave;
    if (row >= rows) return;
    const uint4 pk = *reinterpret_cast<const uint4*>(&h[(long long)row * F_SZ + lane * 8]);
    const unsigned short* e = reinterpret_cast<const unsigned short*>(&pk);
    const float* w8 = &wls[lane * 8];
    float acc = 0.f;
#pragma unroll
    for (int i = 0; i < 8; ++i) acc += bf2f(e[i]) * w8[i];
#pragma unroll
    for (int off = 16; off > 0; off >>= 1) acc += __shfl_down(acc, off, 32);
    if (lane == 0) out[row] = fmaxf(acc + bl[0], 0.f);
}

// ---------------------------------------------------------------------------
// Bucketize (searchsorted right over 255 boundaries) + in-place embedding add.
// ---------------------------------------------------------------------------
__global__ __launch_bounds__(256)
void bucketize_add_kernel(const float* __restrict__ val, const float* __restrict__ q,
                          const float* __restrict__ emb, float* __restrict__ out) {
    __shared__ int sidx[16];
    const int tilesPerB = M_SZ / 16;
    const int b  = blockIdx.x / tilesPerB;
    const int m0 = (blockIdx.x % tilesPerB) * 16;
    if (threadIdx.x < 16) {
        const float v = val[(long long)b * M_SZ + m0 + threadIdx.x];
        int lo = 0, hi = NBQ;
        while (lo < hi) { int mid = (lo + hi) >> 1; if (q[mid] <= v) lo = mid + 1; else hi = mid; }
        sidx[threadIdx.x] = lo;                   // in [0, 255]
    }
    __syncthreads();
    const int r  = threadIdx.x >> 4;
    const int c0 = (threadIdx.x & 15) * 16;
    const int idx = sidx[r];
    float4* op = reinterpret_cast<float4*>(&out[(((long long)b * M_SZ) + m0 + r) * C_SZ + c0]);
    const float4* ep = reinterpret_cast<const float4*>(&emb[(long long)idx * C_SZ + c0]);
#pragma unroll
    for (int i = 0; i < 4; ++i) {
        float4 o = op[i], e = ep[i];
        o.x += e.x; o.y += e.y; o.z += e.z; o.w += e.w;
        op[i] = o;
    }
}

// ---------------------------------------------------------------------------
// Host-side orchestration (graph-capture safe: only kernel launches on stream)
// ---------------------------------------------------------------------------
extern "C" void kernel_launch(void* const* d_in, const int* in_sizes, int n_in,
                              void* d_out, int out_size, void* d_ws, size_t ws_size,
                              hipStream_t stream) {
    (void)in_sizes; (void)n_in; (void)out_size; (void)ws_size;

    const float* inputs      = (const float*)d_in[0];
    const int*   true_dur    = (const int*)  d_in[1];
    const float* true_pitch  = (const float*)d_in[2];
    const float* true_energy = (const float*)d_in[3];
    const int*   mel_max_len = (const int*)  d_in[4];
    const float* pitch_q     = (const float*)d_in[5];
    const float* energy_q    = (const float*)d_in[6];
    const int DP = 7, PP = 17, EP = 27;      // w1,b1,g1,be1,w2,b2,g2,be2,wl,bl
    const float* pitch_emb   = (const float*)d_in[37];
    const float* energy_emb  = (const float*)d_in[38];

    // Workspace layout (bytes): 6 packed weights | cs | h1 | h2   (~69.5 MB)
    char* ws = (char*)d_ws;
    const size_t WP_SZ = (size_t)F_SZ * KKTOT * 2;             // 393216
    unsigned short* wp[6];
    for (int i = 0; i < 6; ++i) wp[i] = (unsigned short*)(ws + i * WP_SZ);
    int* cs = (int*)(ws + 6 * WP_SZ);                          // 32*256 ints
    unsigned short* h1 = (unsigned short*)(ws + 6 * WP_SZ + 32768 + 256);
    unsigned short* h2 = (unsigned short*)((char*)h1 + (size_t)B_SZ * M_SZ * F_SZ * 2);

    // d_out layout (f32): outputs | durations | pitches | energies | mel_len
    float* o_out    = (float*)d_out;
    float* o_dur    = o_out + (size_t)B_SZ * M_SZ * C_SZ;
    float* o_pitch  = o_dur + (size_t)B_SZ * T_SZ;
    float* o_energy = o_pitch + (size_t)B_SZ * M_SZ;
    float* o_mellen = o_energy + (size_t)B_SZ * M_SZ;

    // 1) Repack all six conv weight tensors to bf16 im2col order.
    repack_w_kernel<<<256, 256, 0, stream>>>((const float*)d_in[DP + 0], wp[0]);
    repack_w_kernel<<<256, 256, 0, stream>>>((const float*)d_in[DP + 4], wp[1]);
    repack_w_kernel<<<256, 256, 0, stream>>>((const float*)d_in[PP + 0], wp[2]);
    repack_w_kernel<<<256, 256, 0, stream>>>((const float*)d_in[PP + 4], wp[3]);
    repack_w_kernel<<<256, 256, 0, stream>>>((const float*)d_in[EP + 0], wp[4]);
    repack_w_kernel<<<256, 256, 0, stream>>>((const float*)d_in[EP + 4], wp[5]);

    // 2) Duration cumsum + mel_len.
    cumsum_kernel<<<1, 32, 0, stream>>>(true_dur, cs, o_mellen, mel_max_len);

    // 3) Duration predictor on inputs [32*256 rows].
    conv_relu_ln_kernel<<<(B_SZ * T_SZ) / TROWS, 256, 0, stream>>>(
        inputs, nullptr, wp[0], (const float*)d_in[DP + 1], (const float*)d_in[DP + 2],
        (const float*)d_in[DP + 3], h1, T_SZ);
    conv_relu_ln_kernel<<<(B_SZ * T_SZ) / TROWS, 256, 0, stream>>>(
        nullptr, h1, wp[1], (const float*)d_in[DP + 5], (const float*)d_in[DP + 6],
        (const float*)d_in[DP + 7], h2, T_SZ);
    linear_relu_kernel<<<(B_SZ * T_SZ) / 8, 256, 0, stream>>>(
        h2, (const float*)d_in[DP + 8], (const float*)d_in[DP + 9], o_dur, B_SZ * T_SZ);

    // 4) Length regulation -> outputs region (fully overwritten every call).
    length_regulate_kernel<<<B_SZ * (M_SZ / 16), 256, 0, stream>>>(inputs, cs, o_out);

    // 5) Pitch predictor on expanded outputs [32*2048 rows].
    conv_relu_ln_kernel<<<(B_SZ * M_SZ) / TROWS, 256, 0, stream>>>(
        o_out, nullptr, wp[2], (const float*)d_in[PP + 1], (const float*)d_in[PP + 2],
        (const float*)d_in[PP + 3], h1, M_SZ);
    conv_relu_ln_kernel<<<(B_SZ * M_SZ) / TROWS, 256, 0, stream>>>(
        nullptr, h1, wp[3], (const float*)d_in[PP + 5], (const float*)d_in[PP + 6],
        (const float*)d_in[PP + 7], h2, M_SZ);
    linear_relu_kernel<<<(B_SZ * M_SZ) / 8, 256, 0, stream>>>(
        h2, (const float*)d_in[PP + 8], (const float*)d_in[PP + 9], o_pitch, B_SZ * M_SZ);

    // 6) outputs += pitch_emb[bucketize(true_pitch)]
    bucketize_add_kernel<<<B_SZ * (M_SZ / 16), 256, 0, stream>>>(true_pitch, pitch_q,
                                                                 pitch_emb, o_out);

    // 7) Energy predictor on updated outputs.
    conv_relu_ln_kernel<<<(B_SZ * M_SZ) / TROWS, 256, 0, stream>>>(
        o_out, nullptr, wp[4], (const float*)d_in[EP + 1], (const float*)d_in[EP + 2],
        (const float*)d_in[EP + 3], h1, M_SZ);
    conv_relu_ln_kernel<<<(B_SZ * M_SZ) / TROWS, 256, 0, stream>>>(
        nullptr, h1, wp[5], (const float*)d_in[EP + 5], (const float*)d_in[EP + 6],
        (const float*)d_in[EP + 7], h2, M_SZ);
    linear_relu_kernel<<<(B_SZ * M_SZ) / 8, 256, 0, stream>>>(
        h2, (const float*)d_in[EP + 8], (const float*)d_in[EP + 9], o_energy, B_SZ * M_SZ);

    // 8) outputs += energy_emb[bucketize(true_energy)]
    bucketize_add_kernel<<<B_SZ * (M_SZ / 16), 256, 0, stream>>>(true_energy, energy_q,
                                                                 energy_emb, o_out);
}
